// SparseAttention_46969762349725
// MI455X (gfx1250) — matrix operand order repaired and checked by reading, not compiled, not run on hardware
//
#include <hip/hip_runtime.h>

#define HEADS 8
#define IN_C 256
#define MDL 512
#define NB 4
#define LN 4096
#define NEDGE 65536
#define EDIM 64  // MDL / HEADS

typedef __bf16 bf16x16 __attribute__((ext_vector_type(16)));
typedef float f32x8 __attribute__((ext_vector_type(8)));

// order-preserving float<->u32 encoding for unsigned atomicMax
static __device__ __forceinline__ unsigned encf(float f) {
    unsigned u = __float_as_uint(f);
    return (u & 0x80000000u) ? ~u : (u | 0x80000000u);
}
static __device__ __forceinline__ float decf(unsigned e) {
    unsigned u = (e & 0x80000000u) ? (e & 0x7FFFFFFFu) : ~e;
    return __uint_as_float(u);
}

// -----------------------------------------------------------------------------
// Tiled GEMM:  C[M,512] = A[M,K] @ W[K,512] + bias   (fp32 in/out, bf16 WMMA)
// 256 threads (8 waves), 128x64 tile, BK=32, double-buffered LDS with a
// 2-phase unrolled pipeline (compile-time buffer index, one barrier/K-step).
// Wave w: rows (w&3)*32..+32, cols (w>>2)*32..+32 -> four 16x16 accumulators.
// -----------------------------------------------------------------------------
__global__ __launch_bounds__(256)
void gemm_bias_wmma(const float* __restrict__ A, const float* __restrict__ W,
                    const float* __restrict__ bias, float* __restrict__ C, int K) {
    __shared__ __align__(16) __bf16 sA[2][128 * 40];  // [buf][row][k], stride 40
    __shared__ __align__(16) __bf16 sB[2][64 * 40];   // [buf][col][k], stride 40

    const int t = threadIdx.x;
    const int mBase = blockIdx.y * 128;
    const int nBase = blockIdx.x * 64;

    const int wave = t >> 5;
    const int lane = t & 31;
    const int wm = (wave & 3) * 32;    // wave row base inside tile
    const int wn = (wave >> 2) * 32;   // wave col base inside tile
    const int lrow = lane & 15;
    const int lgrp = lane >> 4;        // half-wave group

    // staging coordinates (branchless)
    const int ar = t >> 3;             // A row slice base (0..31), +32 per j
    const int ac = (t & 7) * 4;        // A col (k) within 32
    const int bk2 = (t >> 4) * 2;      // B k pair base (0,2,..,30)
    const int bn = (t & 15) * 4;       // B col within 64

    const float* aPtr = A + (size_t)(mBase + ar) * K + ac;
    const float* bPtr = W + (size_t)bk2 * MDL + nBase + bn;

    f32x8 acc00 = {}, acc01 = {}, acc10 = {}, acc11 = {};

    float4 ra[4];
    union { float4 v; float f[4]; } rb0, rb1;

    auto loadRegs = [&](int kt) {
        const float* aN = aPtr + kt;
        const float* bN = bPtr + (size_t)kt * MDL;
        #pragma unroll
        for (int j = 0; j < 4; ++j) ra[j] = *(const float4*)(aN + (size_t)j * 32 * K);
        rb0.v = *(const float4*)(bN);
        rb1.v = *(const float4*)(bN + MDL);
    };

    // direct adjacent element stores: backend folds into v_cvt_pk_bf16_f32 +
    // merged 32/64-bit ds stores (do NOT hand-pack; it breaks the folding)
    auto stage = [&](__bf16* dA, __bf16* dB) {
        #pragma unroll
        for (int j = 0; j < 4; ++j) {
            __bf16* p = &dA[(ar + j * 32) * 40 + ac];
            p[0] = (__bf16)ra[j].x; p[1] = (__bf16)ra[j].y;
            p[2] = (__bf16)ra[j].z; p[3] = (__bf16)ra[j].w;
        }
        #pragma unroll
        for (int c = 0; c < 4; ++c) {
            __bf16* p = &dB[(bn + c) * 40 + bk2];
            p[0] = (__bf16)rb0.f[c];
            p[1] = (__bf16)rb1.f[c];
        }
    };

    auto compute = [&](const __bf16* cA, const __bf16* cB) {
        union { bf16x16 v; uint4 q[2]; } fa0, fa1, fb0, fb1;
        const int arow0 = (wm + lrow) * 40;
        const int arow1 = (wm + 16 + lrow) * 40;
        fa0.q[0] = *(const uint4*)&cA[arow0 + lgrp * 8];
        fa0.q[1] = *(const uint4*)&cA[arow0 + 16 + lgrp * 8];
        fa1.q[0] = *(const uint4*)&cA[arow1 + lgrp * 8];
        fa1.q[1] = *(const uint4*)&cA[arow1 + 16 + lgrp * 8];
        const int bcol0 = (wn + lrow) * 40;
        const int bcol1 = (wn + 16 + lrow) * 40;
        fb0.q[0] = *(const uint4*)&cB[bcol0 + lgrp * 8];
        fb0.q[1] = *(const uint4*)&cB[bcol0 + 16 + lgrp * 8];
        fb1.q[0] = *(const uint4*)&cB[bcol1 + lgrp * 8];
        fb1.q[1] = *(const uint4*)&cB[bcol1 + 16 + lgrp * 8];

        acc00 = __builtin_amdgcn_wmma_f32_16x16x32_bf16(false, fa0.v, false, fb0.v,
                                                        (short)0, acc00, false, false);
        acc01 = __builtin_amdgcn_wmma_f32_16x16x32_bf16(false, fa0.v, false, fb1.v,
                                                        (short)0, acc01, false, false);
        acc10 = __builtin_amdgcn_wmma_f32_16x16x32_bf16(false, fa1.v, false, fb0.v,
                                                        (short)0, acc10, false, false);
        acc11 = __builtin_amdgcn_wmma_f32_16x16x32_bf16(false, fa1.v, false, fb1.v,
                                                        (short)0, acc11, false, false);
    };

    // prologue: slab 0 -> buf0
    loadRegs(0);
    stage(sA[0], sB[0]);
    __syncthreads();

    // K/32 is even (K = 256 or 512): exact 2-phase double-buffered pipeline
    for (int kt = 0; kt < K; kt += 64) {
        // phase 0: compute buf0 (slab kt), stage slab kt+32 -> buf1
        loadRegs(kt + 32);
        compute(sA[0], sB[0]);
        stage(sA[1], sB[1]);
        __syncthreads();

        // phase 1: compute buf1 (slab kt+32), stage slab kt+64 -> buf0
        const bool more = (kt + 64) < K;
        if (more) loadRegs(kt + 64);
        compute(sA[1], sB[1]);
        if (more) stage(sA[0], sB[0]);
        __syncthreads();
    }

    // C/D layout: VGPR r -> row M = r + 8*lgrp, col N = lrow
    const int col0 = nBase + wn + lrow;
    const int col1 = col0 + 16;
    const float b0 = bias[col0];
    const float b1 = bias[col1];
    union { f32x8 v; float f[8]; } u00, u01, u10, u11;
    u00.v = acc00; u01.v = acc01; u10.v = acc10; u11.v = acc11;
    #pragma unroll
    for (int r = 0; r < 8; ++r) {
        const size_t row0 = (size_t)(mBase + wm + lgrp * 8 + r);
        const size_t row1 = row0 + 16;
        C[row0 * MDL + col0] = u00.f[r] + b0;
        C[row0 * MDL + col1] = u01.f[r] + b1;
        C[row1 * MDL + col0] = u10.f[r] + b0;
        C[row1 * MDL + col1] = u11.f[r] + b1;
    }
}

// -----------------------------------------------------------------------------
// Per-edge scores + segment max.  8 lanes per edge, shfl_xor reduction.
// -----------------------------------------------------------------------------
__global__ __launch_bounds__(256)
void scores_kernel(const float* __restrict__ qp, const float* __restrict__ kp,
                   const int* __restrict__ dstv, const int* __restrict__ srcv,
                   const float* __restrict__ epe, float* __restrict__ scores,
                   unsigned* __restrict__ menc) {
    const int t = threadIdx.x;
    const int sub = t & 7;
    const int e = blockIdx.x * 32 + (t >> 3);
    const int nh = blockIdx.y;
    const int n = nh >> 3, h = nh & 7;
    const int dst = dstv[e], src = srcv[e];
    const float4* qr = (const float4*)(qp + ((size_t)(n * LN + dst) * MDL + h * EDIM));
    const float4* kr = (const float4*)(kp + ((size_t)(n * LN + src) * MDL + h * EDIM));
    const float4 a0 = qr[sub], a1 = qr[sub + 8];
    const float4 b0 = kr[sub], b1 = kr[sub + 8];
    float acc = a0.x * b0.x + a0.y * b0.y + a0.z * b0.z + a0.w * b0.w
              + a1.x * b1.x + a1.y * b1.y + a1.z * b1.z + a1.w * b1.w;
    acc += __shfl_xor(acc, 1);
    acc += __shfl_xor(acc, 2);
    acc += __shfl_xor(acc, 4);
    if (sub == 0) {
        const float s = 0.125f * (acc + epe[(size_t)nh * NEDGE + e]);  // 1/sqrt(64)
        scores[(size_t)nh * NEDGE + e] = s;
        atomicMax(&menc[nh * LN + dst], encf(s));
    }
}

// -----------------------------------------------------------------------------
// p = exp(s - m[dst]); z[dst] += p   (scores buffer overwritten with p)
// -----------------------------------------------------------------------------
__global__ __launch_bounds__(256)
void expz_kernel(const int* __restrict__ dstv, float* __restrict__ scores,
                 const unsigned* __restrict__ menc, float* __restrict__ z) {
    const int e = blockIdx.x * 256 + threadIdx.x;
    const int nh = blockIdx.y;
    const int dst = dstv[e];
    const float m = decf(menc[nh * LN + dst]);
    const float p = __expf(scores[(size_t)nh * NEDGE + e] - m);
    scores[(size_t)nh * NEDGE + e] = p;
    unsafeAtomicAdd(&z[nh * LN + dst], p);
}

// -----------------------------------------------------------------------------
// attn[n, dst, h*64+d] += (p/z[dst]) * v[n, src, h*64+d]; one wave per edge,
// float2 per lane (coalesced v reads).
// -----------------------------------------------------------------------------
__global__ __launch_bounds__(256)
void agg_kernel(const int* __restrict__ dstv, const int* __restrict__ srcv,
                const float* __restrict__ p, const float* __restrict__ z,
                const float* __restrict__ vp, float* __restrict__ attn) {
    const int t = blockIdx.x * 256 + threadIdx.x;
    const int e = t >> 5;
    const int d0 = (t & 31) * 2;
    const int nh = blockIdx.y;
    const int n = nh >> 3, h = nh & 7;
    const int dst = dstv[e], src = srcv[e];
    const float alpha = p[(size_t)nh * NEDGE + e] / z[nh * LN + dst];
    const float2 v2 = *(const float2*)(vp + (size_t)(n * LN + src) * MDL + h * EDIM + d0);
    float* o = attn + (size_t)(n * LN + dst) * MDL + h * EDIM + d0;
    unsafeAtomicAdd(o + 0, alpha * v2.x);
    unsafeAtomicAdd(o + 1, alpha * v2.y);
}

extern "C" void kernel_launch(void* const* d_in, const int* in_sizes, int n_in,
                              void* d_out, int out_size, void* d_ws, size_t ws_size,
                              hipStream_t stream) {
    const float* queries = (const float*)d_in[0];
    const float* keys    = (const float*)d_in[1];
    const float* values  = (const float*)d_in[2];
    const int*   adj     = (const int*)d_in[3];
    const float* epe     = (const float*)d_in[4];
    const float* Wq      = (const float*)d_in[5];
    const float* bq      = (const float*)d_in[6];
    const float* Wk      = (const float*)d_in[7];
    const float* bk      = (const float*)d_in[8];
    const float* Wv      = (const float*)d_in[9];
    const float* bv      = (const float*)d_in[10];
    const float* Wo      = (const float*)d_in[11];
    const float* bo      = (const float*)d_in[12];
    float* out = (float*)d_out;

    const size_t MROWS = (size_t)NB * LN;  // 16384
    float*    qp     = (float*)d_ws;
    float*    kp     = qp + MROWS * MDL;
    float*    vp     = kp + MROWS * MDL;
    float*    scores = vp + MROWS * MDL;
    unsigned* menc   = (unsigned*)(scores + (size_t)NB * HEADS * NEDGE);
    float*    zbuf   = (float*)(menc + (size_t)NB * HEADS * LN);
    float*    attn   = zbuf + (size_t)NB * HEADS * LN;

    hipMemsetAsync(menc, 0, (size_t)NB * HEADS * LN * sizeof(unsigned), stream);
    hipMemsetAsync(zbuf, 0, (size_t)NB * HEADS * LN * sizeof(float), stream);
    hipMemsetAsync(attn, 0, MROWS * MDL * sizeof(float), stream);

    const dim3 blk(256);
    const dim3 gProj(MDL / 64, (unsigned)(MROWS / 128));  // (8, 128)

    gemm_bias_wmma<<<gProj, blk, 0, stream>>>(queries, Wq, bq, qp, IN_C);
    gemm_bias_wmma<<<gProj, blk, 0, stream>>>(keys,    Wk, bk, kp, IN_C);
    gemm_bias_wmma<<<gProj, blk, 0, stream>>>(values,  Wv, bv, vp, IN_C);

    const dim3 gS(NEDGE / 32, NB * HEADS);
    scores_kernel<<<gS, blk, 0, stream>>>(qp, kp, adj, adj + NEDGE, epe, scores, menc);

    const dim3 gE(NEDGE / 256, NB * HEADS);
    expz_kernel<<<gE, blk, 0, stream>>>(adj, scores, menc, zbuf);

    const dim3 gA((NEDGE * 32) / 256, NB * HEADS);
    agg_kernel<<<gA, blk, 0, stream>>>(adj, adj + NEDGE, scores, zbuf, vp, attn);

    gemm_bias_wmma<<<gProj, blk, 0, stream>>>(attn, Wo, bo, out, MDL);
}
